// StabilizedAttention_84198538870847
// MI455X (gfx1250) — matrix operand order repaired and checked by reading, not compiled, not run on hardware
//
#include <hip/hip_runtime.h>
#include <hip/hip_bf16.h>

typedef __bf16 bf16_t;
typedef __attribute__((ext_vector_type(16))) __bf16 bf16x16;
typedef __attribute__((ext_vector_type(8)))  __bf16 bf16x8;
typedef __attribute__((ext_vector_type(8)))  float  f32x8;
typedef __attribute__((ext_vector_type(4)))  int    i32x4;

#define B_SZ 4
#define T_SZ 2048
#define C_SZ 1024
#define H_SZ 16
#define D_SZ 64
#define M_ROWS (B_SZ * T_SZ)   /* 8192 */
#define N_QKV (3 * C_SZ)       /* 3072 */
#define K_DIM C_SZ             /* 1024 */

#define WMMA_BF16(A, Bm, Cm) \
  __builtin_amdgcn_wmma_f32_16x16x32_bf16(false, (A), false, (Bm), (short)0, (Cm), false, false)

// ---------------------------------------------------------------------------
// Async global->LDS copy (CDNA5 GLOBAL_LOAD_ASYNC_TO_LDS_B128, ASYNCcnt).
// Probe-confirmed: builtin exists here with generic int4* pointer params.
// Fallback keeps the file compiling on toolchains without the builtin.
// ---------------------------------------------------------------------------
#ifndef __has_builtin
#define __has_builtin(x) 0
#endif

#if __has_builtin(__builtin_amdgcn_global_load_async_to_lds_b128)
#define HAVE_ASYNC_LDS 1
#define ASYNC_CP_B128(g, l) \
  __builtin_amdgcn_global_load_async_to_lds_b128((i32x4*)(g), (i32x4*)(l), 0, 0)
#else
#define HAVE_ASYNC_LDS 0
#define ASYNC_CP_B128(g, l) (*(bf16x8*)(l) = *(const bf16x8*)(g))
#endif

#if HAVE_ASYNC_LDS
#if __has_builtin(__builtin_amdgcn_s_wait_asynccnt)
#define WAIT_ASYNC(n) __builtin_amdgcn_s_wait_asynccnt(n)
#else
#define WAIT_ASYNC(n) asm volatile("s_wait_asynccnt %0" ::"i"(n))
#endif
#else
#define WAIT_ASYNC(n) ((void)0)
#endif

// ---------------------------------------------------------------------------
// A-fragment builder from a row pointer (row-major, 32 k-elems visible).
// This code's contiguous k order: lanes 0-15 hold row M=lane, K = {0..15};
//                 lanes 16-31 hold row M=lane-16, K = {16..31}.
// ---------------------------------------------------------------------------
__device__ __forceinline__ bf16x16 build_a_frag(const bf16_t* rowk, int lane) {
  const int hi = (lane >= 16) ? 16 : 0;
  bf16x8 lo = *(const bf16x8*)(rowk + hi);
  bf16x8 up = *(const bf16x8*)(rowk + 8 + hi);
  bf16x16 a;
#pragma unroll
  for (int i = 0; i < 8; ++i) { a[i] = lo[i]; a[8 + i] = up[i]; }
  return a;
}

// ---------------------------------------------------------------------------
// Elementwise fp32 -> bf16 (4 elems/thread)
// ---------------------------------------------------------------------------
__global__ __launch_bounds__(256) void cvt_f32_bf16_kernel(
    const float* __restrict__ in, bf16_t* __restrict__ out, int n) {
  int idx = (blockIdx.x * blockDim.x + threadIdx.x) * 4;
  if (idx + 3 < n) {
    float4 v = *(const float4*)(in + idx);
    out[idx + 0] = (bf16_t)v.x;
    out[idx + 1] = (bf16_t)v.y;
    out[idx + 2] = (bf16_t)v.z;
    out[idx + 3] = (bf16_t)v.w;
  }
}

// ---------------------------------------------------------------------------
// fp32 [K,N] -> bf16 transposed [N,K]
// ---------------------------------------------------------------------------
__global__ __launch_bounds__(256) void cvt_transpose_kernel(
    const float* __restrict__ w, bf16_t* __restrict__ wT, int K, int N) {
  int i = blockIdx.x * blockDim.x + threadIdx.x;
  if (i < K * N) {
    int k = i / N, n = i % N;
    wT[n * K + k] = (bf16_t)w[i];
  }
}

// ---------------------------------------------------------------------------
// Stage one k-step tile set into LDS: A 64x32 (4KB, 1 b128/thread),
// B 256x32 (16KB, 4 b128/thread). 5 async ops per thread per tile.
// ---------------------------------------------------------------------------
__device__ __forceinline__ void stage_tile(
    const bf16_t* __restrict__ A, const bf16_t* __restrict__ Bt,
    int tm0, int tn0, int k0, int tid, bf16_t* ldsA, bf16_t* ldsB) {
  {
    int row = tid >> 2, ch = (tid & 3) * 8;
    ASYNC_CP_B128(A + (tm0 + row) * K_DIM + k0 + ch, ldsA + row * 32 + ch);
  }
#pragma unroll
  for (int q = 0; q < 4; ++q) {
    int idx = tid + (q << 8);
    int row = idx >> 2, ch = (idx & 3) * 8;
    ASYNC_CP_B128(Bt + (tn0 + row) * K_DIM + k0 + ch, ldsB + row * 32 + ch);
  }
}

// ---------------------------------------------------------------------------
// Block GEMM body: 8 waves (2M x 4N), block tile 64M x 256N, LDS-staged,
// double-buffered async pipeline, 8 WMMAs per wave per 32-k step.
// acc must be zero-initialized by caller.
// ---------------------------------------------------------------------------
__device__ __forceinline__ void gemm_block_64x256(
    const bf16_t* __restrict__ A, const bf16_t* __restrict__ Bt,
    int tm0, int tn0, bf16_t* ldsA, bf16_t* ldsB,  // [2][...] buffers
    f32x8 acc[2][4]) {
  const int tid  = threadIdx.x;
  const int lane = tid & 31;
  const int wave = tid >> 5;
  const int wm = wave >> 2, wn = wave & 3;

  const int hi16 = (lane >= 16) ? 16 : 0;
  const int lrA0 = (wm * 32 + (lane & 15)) * 32;
  const int lrA1 = (wm * 32 + 16 + (lane & 15)) * 32;
  const int lcB  = (wn * 64 + (lane & 15)) * 32;

  stage_tile(A, Bt, tm0, tn0, 0, tid, ldsA, ldsB);

  const int NK = K_DIM / 32;
  for (int kt = 0; kt < NK; ++kt) {
    const int cur = kt & 1, nxt = cur ^ 1;
    if (kt + 1 < NK) {
      stage_tile(A, Bt, tm0, tn0, (kt + 1) * 32, tid,
                 ldsA + nxt * (64 * 32), ldsB + nxt * (256 * 32));
      WAIT_ASYNC(5);   // oldest 5 (tile kt) complete; tile kt+1 in flight
    } else {
      WAIT_ASYNC(0);
    }
    __syncthreads();   // all waves' tile-kt data visible in LDS

    const bf16_t* tA = ldsA + cur * (64 * 32);
    const bf16_t* tB = ldsB + cur * (256 * 32);

    bf16x16 a0 = build_a_frag(tA + lrA0, lane);
    bf16x16 a1 = build_a_frag(tA + lrA1, lane);
    bf16x16 b0 = *(const bf16x16*)(tB + lcB + (0 * 16) * 32 + hi16);
    bf16x16 b1 = *(const bf16x16*)(tB + lcB + (1 * 16) * 32 + hi16);
    bf16x16 b2 = *(const bf16x16*)(tB + lcB + (2 * 16) * 32 + hi16);
    bf16x16 b3 = *(const bf16x16*)(tB + lcB + (3 * 16) * 32 + hi16);

    acc[0][0] = WMMA_BF16(a0, b0, acc[0][0]);
    acc[0][1] = WMMA_BF16(a0, b1, acc[0][1]);
    acc[0][2] = WMMA_BF16(a0, b2, acc[0][2]);
    acc[0][3] = WMMA_BF16(a0, b3, acc[0][3]);
    acc[1][0] = WMMA_BF16(a1, b0, acc[1][0]);
    acc[1][1] = WMMA_BF16(a1, b1, acc[1][1]);
    acc[1][2] = WMMA_BF16(a1, b2, acc[1][2]);
    acc[1][3] = WMMA_BF16(a1, b3, acc[1][3]);

    __syncthreads();   // readers done before buffer is re-staged
  }
}

// ---------------------------------------------------------------------------
// QKV GEMM: qkv = x @ w_qkv + b_qkv (M=8192, N=3072, K=1024).
// Scatter-stores q,k -> [B,H,T,D] and v -> [B,H,D,T] in bf16.
// ---------------------------------------------------------------------------
__global__ __launch_bounds__(256) void qkv_gemm_kernel(
    const bf16_t* __restrict__ xb,    // [8192,1024]
    const bf16_t* __restrict__ wT,    // [3072,1024]
    const float*  __restrict__ bias,  // [3072]
    bf16_t* __restrict__ qc,          // [B,H,T,D]
    bf16_t* __restrict__ kc,          // [B,H,T,D]
    bf16_t* __restrict__ vT) {        // [B,H,D,T]
  __shared__ bf16_t ldsA[2][64 * 32];
  __shared__ bf16_t ldsB[2][256 * 32];
  const int lane = threadIdx.x & 31;
  const int wave = threadIdx.x >> 5;
  const int tm0 = blockIdx.y * 64, tn0 = blockIdx.x * 256;
  const int tm = tm0 + (wave >> 2) * 32;
  const int tn = tn0 + (wave & 3) * 64;

  f32x8 acc[2][4] = {};
  gemm_block_64x256(xb, wT, tm0, tn0, &ldsA[0][0], &ldsB[0][0], acc);

  const int rowOff = (lane >= 16) ? 8 : 0;
#pragma unroll
  for (int j = 0; j < 4; ++j) {
    const int col = tn + j * 16 + (lane & 15);
    const float bv = bias[col];
    const int seg = col / C_SZ;  // 0=q, 1=k, 2=v
    const int nin = col % C_SZ;
    const int h = nin / D_SZ;
    const int d = nin % D_SZ;
#pragma unroll
    for (int mi = 0; mi < 2; ++mi) {
#pragma unroll
      for (int i = 0; i < 8; ++i) {
        int r  = tm + mi * 16 + i + rowOff;
        int bb = r / T_SZ;
        int t  = r % T_SZ;
        bf16_t val = (bf16_t)(acc[mi][j][i] + bv);
        if (seg == 0)
          qc[((bb * H_SZ + h) * T_SZ + t) * D_SZ + d] = val;
        else if (seg == 1)
          kc[((bb * H_SZ + h) * T_SZ + t) * D_SZ + d] = val;
        else
          vT[((bb * H_SZ + h) * D_SZ + d) * T_SZ + t] = val;
      }
    }
  }
}

// ---------------------------------------------------------------------------
// Flash attention: one wave per (b,h,q-tile of 16). Online softmax; causal.
// Scores via 4 WMMAs per 32-k step, PV via 4 WMMAs against transposed V.
// P re-shaped C-layout -> A-layout through LDS.
// ---------------------------------------------------------------------------
__global__ __launch_bounds__(128) void attn_kernel(
    const bf16_t* __restrict__ qc,   // [B,H,T,D]
    const bf16_t* __restrict__ kc,   // [B,H,T,D]
    const bf16_t* __restrict__ vT,   // [B,H,D,T]
    bf16_t* __restrict__ attn) {     // [8192,1024]
  __shared__ bf16_t pShared[4][16 * 32];
  const int lane = threadIdx.x & 31;
  const int wave = threadIdx.x >> 5;
  const int wid  = blockIdx.x * 4 + wave;  // 0..8191
  const int qt   = wid & 127;
  const int bh   = wid >> 7;               // b*16 + h
  const int q0   = qt * 16;

  const bf16_t* qhead = qc + bh * (T_SZ * D_SZ);
  const bf16_t* khead = kc + bh * (T_SZ * D_SZ);
  const bf16_t* vhead = vT + bh * (D_SZ * T_SZ);
  bf16_t* myP = pShared[wave];

  const int qrowA = q0 + (lane & 15);
  bf16x16 aq0 = build_a_frag(qhead + qrowA * D_SZ + 0, lane);
  bf16x16 aq1 = build_a_frag(qhead + qrowA * D_SZ + 32, lane);

  f32x8 o0 = {}, o1 = {}, o2 = {}, o3 = {};
  float mrow[8], lrow[8];
#pragma unroll
  for (int i = 0; i < 8; ++i) { mrow[i] = -1e30f; lrow[i] = 0.f; }

  const int rowOff = (lane >= 16) ? 8 : 0;
  const int hi16   = (lane >= 16) ? 16 : 0;
  const float scale = 0.125f;  // 1/sqrt(64)

  for (int k0 = 0; k0 <= q0 + 15; k0 += 32) {
    f32x8 s[2];
#pragma unroll
    for (int sub = 0; sub < 2; ++sub) {
      const int kcol = k0 + sub * 16 + (lane & 15);
      bf16x16 bk0 = *(const bf16x16*)(khead + kcol * D_SZ + 0 + hi16);
      bf16x16 bk1 = *(const bf16x16*)(khead + kcol * D_SZ + 32 + hi16);
      f32x8 c = {};
      c = WMMA_BF16(aq0, bk0, c);
      c = WMMA_BF16(aq1, bk1, c);
#pragma unroll
      for (int i = 0; i < 8; ++i) {
        int rq = q0 + i + rowOff;
        c[i] = (kcol > rq) ? -1e30f : c[i] * scale;
      }
      s[sub] = c;
    }

    float nm[8], alpha[8];
#pragma unroll
    for (int i = 0; i < 8; ++i) {
      float v = fmaxf(s[0][i], s[1][i]);
#pragma unroll
      for (int msk = 8; msk >= 1; msk >>= 1)
        v = fmaxf(v, __shfl_xor(v, msk, 32));
      nm[i] = fmaxf(mrow[i], v);
      alpha[i] = __expf(mrow[i] - nm[i]);
    }
#pragma unroll
    for (int i = 0; i < 8; ++i) {
      float p0 = __expf(s[0][i] - nm[i]);
      float p1 = __expf(s[1][i] - nm[i]);
      int prow = i + rowOff;
      int pcol = lane & 15;
      myP[prow * 32 + pcol]      = (bf16_t)p0;
      myP[prow * 32 + 16 + pcol] = (bf16_t)p1;
      float v = p0 + p1;
#pragma unroll
      for (int msk = 8; msk >= 1; msk >>= 1)
        v += __shfl_xor(v, msk, 32);
      lrow[i] = lrow[i] * alpha[i] + v;
      mrow[i] = nm[i];
    }

#pragma unroll
    for (int i = 0; i < 8; ++i) {
      o0[i] *= alpha[i]; o1[i] *= alpha[i];
      o2[i] *= alpha[i]; o3[i] *= alpha[i];
    }

    bf16x16 pf;
    {
      int prow = lane & 15;
      int phi  = (lane >= 16) ? 16 : 0;
      bf16x8 lo = *(const bf16x8*)(myP + prow * 32 + phi);
      bf16x8 up = *(const bf16x8*)(myP + prow * 32 + 8 + phi);
#pragma unroll
      for (int i = 0; i < 8; ++i) { pf[i] = lo[i]; pf[8 + i] = up[i]; }
    }

#pragma unroll
    for (int j = 0; j < 4; ++j) {
      bf16x16 bv = *(const bf16x16*)(vhead + (j * 16 + (lane & 15)) * T_SZ +
                                     k0 + hi16);
      f32x8* oj = (j == 0) ? &o0 : (j == 1) ? &o1 : (j == 2) ? &o2 : &o3;
      *oj = WMMA_BF16(pf, bv, *oj);
    }
  }

  const int bI = bh >> 4;
  const int h  = bh & 15;
#pragma unroll
  for (int i = 0; i < 8; ++i) {
    float inv = 1.f / lrow[i];
    int r = bI * T_SZ + q0 + i + rowOff;
    int cb = h * D_SZ + (lane & 15);
    attn[r * C_SZ + cb + 0]  = (bf16_t)(o0[i] * inv);
    attn[r * C_SZ + cb + 16] = (bf16_t)(o1[i] * inv);
    attn[r * C_SZ + cb + 32] = (bf16_t)(o2[i] * inv);
    attn[r * C_SZ + cb + 48] = (bf16_t)(o3[i] * inv);
  }
}

// ---------------------------------------------------------------------------
// Output projection: out = attn @ w_out + b_out (M=8192, N=K=1024), fp32 out.
// ---------------------------------------------------------------------------
__global__ __launch_bounds__(256) void proj_gemm_kernel(
    const bf16_t* __restrict__ ab,    // [8192,1024]
    const bf16_t* __restrict__ wT,    // [1024,1024]
    const float*  __restrict__ bias,  // [1024]
    float* __restrict__ out) {        // [8192,1024]
  __shared__ bf16_t ldsA[2][64 * 32];
  __shared__ bf16_t ldsB[2][256 * 32];
  const int lane = threadIdx.x & 31;
  const int wave = threadIdx.x >> 5;
  const int tm0 = blockIdx.y * 64, tn0 = blockIdx.x * 256;
  const int tm = tm0 + (wave >> 2) * 32;
  const int tn = tn0 + (wave & 3) * 64;

  f32x8 acc[2][4] = {};
  gemm_block_64x256(ab, wT, tm0, tn0, &ldsA[0][0], &ldsB[0][0], acc);

  const int rowOff = (lane >= 16) ? 8 : 0;
#pragma unroll
  for (int j = 0; j < 4; ++j) {
    const int col = tn + j * 16 + (lane & 15);
    const float bv = bias[col];
#pragma unroll
    for (int mi = 0; mi < 2; ++mi) {
#pragma unroll
      for (int i = 0; i < 8; ++i) {
        int r = tm + mi * 16 + i + rowOff;
        out[r * C_SZ + col] = acc[mi][j][i] + bv;
      }
    }
  }
}

// ---------------------------------------------------------------------------
// Launch
// ---------------------------------------------------------------------------
extern "C" void kernel_launch(void* const* d_in, const int* in_sizes, int n_in,
                              void* d_out, int out_size, void* d_ws,
                              size_t ws_size, hipStream_t stream) {
  const float* x     = (const float*)d_in[0];
  const float* w_qkv = (const float*)d_in[1];
  const float* b_qkv = (const float*)d_in[2];
  const float* w_out = (const float*)d_in[3];
  const float* b_out = (const float*)d_in[4];
  float* out = (float*)d_out;

  char* ws = (char*)d_ws;
  bf16_t* xb    = (bf16_t*)(ws + (size_t)0);           // 16 MB
  bf16_t* wqkvT = (bf16_t*)(ws + ((size_t)16 << 20));  //  6 MB
  bf16_t* woutT = (bf16_t*)(ws + ((size_t)22 << 20));  //  2 MB
  bf16_t* qc    = (bf16_t*)(ws + ((size_t)24 << 20));  // 16 MB
  bf16_t* kc    = (bf16_t*)(ws + ((size_t)40 << 20));  // 16 MB
  bf16_t* vT    = (bf16_t*)(ws + ((size_t)56 << 20));  // 16 MB
  bf16_t* attn  = (bf16_t*)(ws + ((size_t)72 << 20));  // 16 MB (total 88 MB)

  // 1. Precision staging
  cvt_f32_bf16_kernel<<<(M_ROWS * C_SZ) / 4 / 256, 256, 0, stream>>>(
      x, xb, M_ROWS * C_SZ);
  cvt_transpose_kernel<<<(C_SZ * N_QKV + 255) / 256, 256, 0, stream>>>(
      w_qkv, wqkvT, C_SZ, N_QKV);
  cvt_transpose_kernel<<<(C_SZ * C_SZ + 255) / 256, 256, 0, stream>>>(
      w_out, woutT, C_SZ, C_SZ);

  // 2. QKV projection (WMMA, async-LDS double-buffered)
  qkv_gemm_kernel<<<dim3(N_QKV / 256, M_ROWS / 64), 256, 0, stream>>>(
      xb, wqkvT, b_qkv, qc, kc, vT);

  // 3. Flash attention (WMMA + LDS)
  attn_kernel<<<(B_SZ * H_SZ * (T_SZ / 16)) / 4, 128, 0, stream>>>(
      qc, kc, vT, attn);

  // 4. Output projection (WMMA, async-LDS double-buffered)
  proj_gemm_kernel<<<dim3(C_SZ / 256, M_ROWS / 64), 256, 0, stream>>>(
      attn, woutT, b_out, out);
}
